// VGG_wfc_67894843015706
// MI455X (gfx1250) — matrix operand order repaired and checked by reading, not compiled, not run on hardware
//
#include <hip/hip_runtime.h>
#include <math.h>

// ---------------------------------------------------------------------------
// Types for CDNA5 WMMA (gfx1250, wave32)
// ---------------------------------------------------------------------------
typedef __attribute__((ext_vector_type(16))) __bf16 v16bf;
typedef __attribute__((ext_vector_type(8)))  float  v8f;

union BF16x16 {
    v16bf v;
    uint4 q[2];
};

static __device__ __forceinline__ unsigned short f32_to_bf16(float f) {
    union { float f; unsigned u; } v;
    v.f = f;
    unsigned u = v.u;
    unsigned r = (u + 0x7FFFu + ((u >> 16) & 1u)) >> 16;   // round-to-nearest-even
    return (unsigned short)r;
}

// ---------------------------------------------------------------------------
// Layer 0: direct conv 3->64 (K=27, too ragged for WMMA), NCHW f32 -> NHWC f32
// ---------------------------------------------------------------------------
__global__ __launch_bounds__(256)
void conv_first(const float* __restrict__ x, const float* __restrict__ w,
                const float* __restrict__ bias, float* __restrict__ out)
{
    int t = blockIdx.x * 256 + threadIdx.x;              // 256*1024*64 threads
    int c   = t & 63;
    int pix = t >> 6;
    int px = pix & 31;
    int py = (pix >> 5) & 31;
    int b  = pix >> 10;
    float acc = bias[c];
    #pragma unroll
    for (int ci = 0; ci < 3; ++ci) {
        #pragma unroll
        for (int ky = 0; ky < 3; ++ky) {
            int iy = py + ky - 1;
            if (iy < 0 || iy > 31) continue;
            #pragma unroll
            for (int kx = 0; kx < 3; ++kx) {
                int ix = px + kx - 1;
                if (ix < 0 || ix > 31) continue;
                acc += x[((b * 3 + ci) * 32 + iy) * 32 + ix] *
                       w[((c * 3 + ci) * 3 + ky) * 3 + kx];
            }
        }
    }
    out[t] = acc;   // NHWC: t == pix*64 + c
}

// ---------------------------------------------------------------------------
// Weight repack: (Cout,Cin,3,3) f32 -> [Cout][ky][kx][Cin] bf16
// grid.y = cout, grid.x covers Cin*9; only constant-divisor math remains.
// ---------------------------------------------------------------------------
__global__ __launch_bounds__(256)
void repack_w(const float* __restrict__ w, unsigned short* __restrict__ dst,
              int Cin)
{
    int co = blockIdx.y;
    int r  = blockIdx.x * 256 + threadIdx.x;             // index within one cout
    if (r >= Cin * 9) return;
    int ci = r / 9;                                      // constant divisor
    int kk = r - ci * 9;
    size_t base = (size_t)co * 9 * Cin;
    dst[base + (size_t)kk * Cin + ci] = f32_to_bf16(w[base + r]);
}

__global__ __launch_bounds__(256)
void repack_fc(const float* __restrict__ w, unsigned short* __restrict__ dst)
{
    int t = blockIdx.x * 256 + threadIdx.x;
    if (t >= 16 * 512) return;
    int n = t >> 9, k = t & 511;
    dst[t] = f32_to_bf16(n < 10 ? w[n * 512 + k] : 0.0f);
}

// ---------------------------------------------------------------------------
// Implicit-GEMM conv with v_wmma_f32_16x16x32_bf16, templated so all index
// math is shifts and the K-loop unrolls. One wave computes a 16(M) x 64(N)
// strip: one A fragment is reused across 4 WMMAs (4 cout tiles).
//   act : bf16 NHWC (256,H,H,Cin)       (Cin multiple of 32)
//   wpk : bf16 [Cout][ky][kx][Cin]
//   out : f32  NHWC (256,H,H,Cout)
// A-fragment (16-bit, 16x32): lanes 0-15 rows M, K{0-7,16-23};
//                             lanes16-31 rows M, K{8-15,24-31}.
// B-fragment (16-bit, 32x16): lanes 0-15 cols N, K{0-7,16-23};
//                             lanes16-31 cols N, K{8-15,24-31}.
// ---------------------------------------------------------------------------
template <int H, int Cin, int Cout>
__global__ __launch_bounds__(256)
void conv_wmma_t(const unsigned short* __restrict__ act,
                 const unsigned short* __restrict__ wpk,
                 const float* __restrict__ bias,
                 float* __restrict__ out)
{
    constexpr int HW = H * H;
    constexpr int M  = 256 * HW;
    constexpr int nGroups = Cout / 64;                 // 4 n-tiles per group
    constexpr long totalWaves = (long)(M / 16) * nGroups;

    const int lane = threadIdx.x & 31;
    const int wave = threadIdx.x >> 5;
    long gw = (long)blockIdx.x * 8 + wave;
    if (gw >= totalWaves) return;
    int mTile  = (int)(gw / nGroups);                  // power-of-two -> shift
    int nGroup = (int)(gw % nGroups);

    const int halfSel = lane >> 4;                     // 0 or 1
    const int l15     = lane & 15;

    // A: this lane's output pixel
    int m   = (mTile << 4) + l15;
    int b   = m / HW;
    int rem = m & (HW - 1);
    int y   = rem / H;
    int x   = rem & (H - 1);

    // B: base column of this group, this lane's column within a tile
    const int n0 = nGroup * 64;
    const unsigned short* wcol =
        wpk + (size_t)(n0 + l15) * 9 * Cin + halfSel * 8;
    constexpr size_t wTileStride = (size_t)16 * 9 * Cin;   // 16 couts

    v8f zero = {};
    v8f acc[4];
    #pragma unroll
    for (int j = 0; j < 4; ++j) acc[j] = zero;

    for (int ky = 0; ky < 3; ++ky) {
        int iy = y + ky - 1;
        for (int kx = 0; kx < 3; ++kx) {
            int ix = x + kx - 1;
            bool valid = ((unsigned)iy < (unsigned)H) && ((unsigned)ix < (unsigned)H);
            const unsigned short* arow =
                act + (size_t)((b * H + iy) * H + ix) * Cin + halfSel * 8;
            const unsigned short* wkk = wcol + (size_t)(ky * 3 + kx) * Cin;
            for (int cc = 0; cc < Cin; cc += 32) {
                BF16x16 a;
                if (valid) {
                    a.q[0] = *(const uint4*)(arow + cc);        // K cc+{0..7}(+8*half)
                    a.q[1] = *(const uint4*)(arow + cc + 16);   // K cc+16+{0..7}
                } else {
                    a.q[0] = make_uint4(0u, 0u, 0u, 0u);
                    a.q[1] = make_uint4(0u, 0u, 0u, 0u);
                }
                #pragma unroll
                for (int j = 0; j < 4; ++j) {
                    BF16x16 bb;
                    const unsigned short* wp = wkk + j * wTileStride + cc;
                    bb.q[0] = *(const uint4*)(wp);              // K cc+{0..7}(+8*half)
                    bb.q[1] = *(const uint4*)(wp + 16);         // K cc+16+{0..7}
                    acc[j] = __builtin_amdgcn_wmma_f32_16x16x32_bf16(
                                 false, a.v, false, bb.v, (short)0, acc[j],
                                 false, false);
                }
            }
        }
    }

    // D layout: VGPR r -> M = r + 8*halfSel, N = l15 (per 16-wide tile)
    int pixBase = (mTile << 4) + (halfSel << 3);
    float* obase = out + (size_t)pixBase * Cout + n0 + l15;
    #pragma unroll
    for (int j = 0; j < 4; ++j) {
        float bv = bias[n0 + j * 16 + l15];
        #pragma unroll
        for (int r = 0; r < 8; ++r)
            obase[(size_t)r * Cout + j * 16] = acc[j][r] + bv;
    }
}

// ---------------------------------------------------------------------------
// BN statistics, stage 1: (C x S) blocks produce per-slice partial sums.
// ---------------------------------------------------------------------------
__global__ __launch_bounds__(256)
void bn_partial(const float* __restrict__ yv, int Nelem, int C,
                float2* __restrict__ partial)
{
    int c = blockIdx.x;
    int s = blockIdx.y;
    int S = gridDim.y;
    int chunk = (Nelem + S - 1) / S;
    int i0 = s * chunk;
    int i1 = min(Nelem, i0 + chunk);
    float su = 0.f, sq = 0.f;
    for (int i = i0 + (int)threadIdx.x; i < i1; i += 256) {
        float v = yv[(size_t)i * C + c];
        su += v; sq += v * v;
    }
    __shared__ float ls[256], lq[256];
    ls[threadIdx.x] = su;
    lq[threadIdx.x] = sq;
    __syncthreads();
    for (int off = 128; off; off >>= 1) {
        if (threadIdx.x < off) {
            ls[threadIdx.x] += ls[threadIdx.x + off];
            lq[threadIdx.x] += lq[threadIdx.x + off];
        }
        __syncthreads();
    }
    if (threadIdx.x == 0)
        partial[s * C + c] = make_float2(ls[0], lq[0]);
}

// Stage 2: combine partials -> (scale, shift) per channel.
__global__ __launch_bounds__(256)
void bn_finalize(const float2* __restrict__ partial, int S, int C, int Nelem,
                 const float* __restrict__ g, const float* __restrict__ bb,
                 float2* __restrict__ stats)
{
    int c = blockIdx.x * 256 + threadIdx.x;
    if (c >= C) return;
    float su = 0.f, sq = 0.f;
    for (int s = 0; s < S; ++s) {
        float2 p = partial[s * C + c];
        su += p.x; sq += p.y;
    }
    float inv  = 1.0f / (float)Nelem;
    float mean = su * inv;
    float var  = sq * inv - mean * mean;
    float sc   = g[c] * rsqrtf(var + 1e-5f);
    stats[c] = make_float2(sc, bb[c] - mean * sc);
}

// ---------------------------------------------------------------------------
// Fused BN + ReLU (+ optional 2x2 maxpool) + f32->bf16, NHWC (square spatial).
// All divisors are powers of two, passed as shift counts.
// ---------------------------------------------------------------------------
__global__ __launch_bounds__(256)
void bn_relu_pool(const float* __restrict__ yv, const float2* __restrict__ stats,
                  unsigned short* __restrict__ out,
                  int lgC, int lgWo, int pool)
{
    const int C  = 1 << lgC;
    const int Wo = 1 << lgWo;                    // output spatial (square)
    size_t total = (size_t)256 << (lgC + 2 * lgWo);
    size_t t = (size_t)blockIdx.x * 256 + threadIdx.x;
    if (t >= total) return;
    int c = (int)(t & (C - 1));
    size_t pix = t >> lgC;
    int xo = (int)(pix & (Wo - 1));
    int yo = (int)((pix >> lgWo) & (Wo - 1));
    int b  = (int)(pix >> (2 * lgWo));
    float2 sc = stats[c];
    float v;
    if (pool) {
        const int W = Wo << 1;                   // input spatial
        size_t base = ((size_t)((b << 1) * Wo + yo) * 2 * W + 2 * xo) * C + c;
        // note: (2*b*W + 2*yo)*W == ((b<<1)*Wo + yo)*2*W  since W = 2*Wo
        size_t rowS = (size_t)W * C;
        float a0 = fmaxf(yv[base]            * sc.x + sc.y, 0.f);
        float a1 = fmaxf(yv[base + C]        * sc.x + sc.y, 0.f);
        float a2 = fmaxf(yv[base + rowS]     * sc.x + sc.y, 0.f);
        float a3 = fmaxf(yv[base + rowS + C] * sc.x + sc.y, 0.f);
        v = fmaxf(fmaxf(a0, a1), fmaxf(a2, a3));
    } else {
        v = fmaxf(yv[t] * sc.x + sc.y, 0.f);
    }
    out[t] = f32_to_bf16(v);
}

// ---------------------------------------------------------------------------
// FC: (256,512) bf16 x (512,16) bf16 -> (256,10) f32 via WMMA (N padded to 16)
// ---------------------------------------------------------------------------
__global__ __launch_bounds__(256)
void fc_wmma(const unsigned short* __restrict__ act,
             const unsigned short* __restrict__ wpk,
             const float* __restrict__ bias, float* __restrict__ out)
{
    int lane = threadIdx.x & 31, wave = threadIdx.x >> 5;
    int gw = blockIdx.x * 8 + wave;
    if (gw >= 16) return;
    int halfSel = lane >> 4, l15 = lane & 15;
    int m = gw * 16 + l15;
    const unsigned short* arow = act + (size_t)m * 512 + halfSel * 8;
    const unsigned short* wrow = wpk + (size_t)l15 * 512 + halfSel * 8;
    v8f c = {};
    #pragma unroll
    for (int c0 = 0; c0 < 512; c0 += 32) {
        BF16x16 a, bv;
        a.q[0]  = *(const uint4*)(arow + c0);
        a.q[1]  = *(const uint4*)(arow + c0 + 16);
        bv.q[0] = *(const uint4*)(wrow + c0);
        bv.q[1] = *(const uint4*)(wrow + c0 + 16);
        c = __builtin_amdgcn_wmma_f32_16x16x32_bf16(
                false, a.v, false, bv.v, (short)0, c, false, false);
    }
    int n = l15;
    if (n < 10) {
        int mBase = gw * 16 + halfSel * 8;
        float bb = bias[n];
        #pragma unroll
        for (int r = 0; r < 8; ++r)
            out[(size_t)(mBase + r) * 10 + n] = c[r] + bb;
    }
}

// ---------------------------------------------------------------------------
// Bandpass: per (b,ch) 32x32 block: fft2 -> fftshift*filt -> 20*log|.| ->
// ifft2 -> abs, channel-flipped store. One 1024-thread block per (b,ch).
// ---------------------------------------------------------------------------
__global__ __launch_bounds__(1024)
void bandpass(const float* __restrict__ x, const float* __restrict__ wgt,
              float* __restrict__ out)
{
    __shared__ float xr[1024], xi[1024], yr[1024], yi[1024];
    int blk = blockIdx.x;
    int ch = blk % 3;
    int b  = blk / 3;
    int t   = threadIdx.x;
    int col = t & 31;        // u (freq) / x (space)
    int row = t >> 5;        // y / v
    const float W0 = 6.283185307179586f / 32.0f;

    xr[t] = x[((b * 3 + ch) * 32 + row) * 32 + col];
    __syncthreads();

    // pass A: row DFT of real input -> yr/yi[y*32+u]
    {
        float sr = 0.f, si = 0.f;
        for (int k = 0; k < 32; ++k) {
            float v = xr[row * 32 + k];
            float s, cn; sincosf(-W0 * (float)(col * k), &s, &cn);
            sr += v * cn; si += v * s;
        }
        yr[t] = sr; yi[t] = si;
    }
    __syncthreads();

    // pass B: column DFT -> F in xr/xi[v*32+u]
    {
        float sr = 0.f, si = 0.f;
        for (int k = 0; k < 32; ++k) {
            float ar = yr[k * 32 + col], ai = yi[k * 32 + col];
            float s, cn; sincosf(-W0 * (float)(row * k), &s, &cn);
            sr += ar * cn - ai * s;
            si += ar * s + ai * cn;
        }
        xr[t] = sr; xi[t] = si;
    }
    __syncthreads();

    // mag with fftshift-ed bandpass filter; result is already ifftshift-ed:
    // m[v][u] = 20*log(|F[v][u]| * |filt[(v+16)%32][(u+16)%32]|)
    {
        float fr = xr[t], fi = xi[t];
        int pf = (row + 16) & 31;
        int qf = (col + 16) & 31;
        float dx = (float)(qf - 16), dy = (float)(pf - 16);
        float r  = sqrtf(dx * dx + dy * dy);
        const float bw = 16.0f / 9.0f;
        float band = floorf(r / bw);
        int bi = (int)band;
        bool inside = (r > band * bw) && (r < (band + 1.0f) * bw) && (bi <= 8);
        int idx = inside ? bi : 9;
        float f = wgt[b * 10 + idx];
        float mag = sqrtf(fr * fr + fi * fi);
        yr[t] = 20.0f * logf(mag * fabsf(f));
    }
    __syncthreads();

    // pass C: row IDFT of real m -> xr/xi[v*32+x]
    {
        float sr = 0.f, si = 0.f;
        for (int k = 0; k < 32; ++k) {
            float v = yr[row * 32 + k];
            float s, cn; sincosf(W0 * (float)(col * k), &s, &cn);
            sr += v * cn; si += v * s;
        }
        xr[t] = sr; xi[t] = si;
    }
    __syncthreads();

    // pass D: column IDFT, scale 1/1024, abs, channel-flipped store
    {
        float sr = 0.f, si = 0.f;
        for (int k = 0; k < 32; ++k) {
            float ar = xr[k * 32 + col], ai = xi[k * 32 + col];
            float s, cn; sincosf(W0 * (float)(row * k), &s, &cn);
            sr += ar * cn - ai * s;
            si += ar * s + ai * cn;
        }
        sr *= (1.0f / 1024.0f);
        si *= (1.0f / 1024.0f);
        out[((b * 3 + (2 - ch)) * 32 + row) * 32 + col] = sqrtf(sr * sr + si * si);
    }
}

// ---------------------------------------------------------------------------
// Host-side conv dispatch (all shapes are compile-time template instances)
// ---------------------------------------------------------------------------
static void launch_conv(int H, int Cin, int Cout,
                        const unsigned short* act, const unsigned short* wpk,
                        const float* bias, float* out, hipStream_t stream)
{
    long waves = ((long)256 * H * H / 16) * (Cout / 64);
    unsigned blocks = (unsigned)((waves + 7) / 8);
    #define CASE(h, ci, co)                                                  \
        if (H == h && Cin == ci && Cout == co) {                             \
            conv_wmma_t<h, ci, co><<<blocks, 256, 0, stream>>>(act, wpk, bias, out); \
            return;                                                          \
        }
    CASE(32,  64,  64)
    CASE(16,  64, 128)
    CASE(16, 128, 128)
    CASE( 8, 128, 256)
    CASE( 8, 256, 256)
    CASE( 4, 256, 512)
    CASE( 4, 512, 512)
    CASE( 2, 512, 512)
    #undef CASE
}

static int ilog2(int v) { int r = 0; while ((1 << r) < v) ++r; return r; }

// ---------------------------------------------------------------------------
// Host orchestration
// ---------------------------------------------------------------------------
extern "C" void kernel_launch(void* const* d_in, const int* in_sizes, int n_in,
                              void* d_out, int out_size, void* d_ws, size_t ws_size,
                              hipStream_t stream)
{
    (void)in_sizes; (void)n_in; (void)out_size; (void)ws_size;

    const float* x = (const float*)d_in[0];
    const float* convw[16]; const float* convb[16];
    const float* bng[16];   const float* bnb[16];
    for (int i = 0; i < 16; ++i) {
        convw[i] = (const float*)d_in[1 + i];
        convb[i] = (const float*)d_in[17 + i];
        bng[i]   = (const float*)d_in[33 + i];
        bnb[i]   = (const float*)d_in[49 + i];
    }
    const float* fcw = (const float*)d_in[65];
    const float* fcb = (const float*)d_in[66];

    // workspace layout
    char* ws = (char*)d_ws;
    float*          convout = (float*)ws;                               //  64 MiB f32 conv out
    unsigned short* actA  = (unsigned short*)(ws + 67108864);           //  32 MiB bf16
    unsigned short* actB  = (unsigned short*)(ws + 100663296);          //  32 MiB bf16
    unsigned short* wpk   = (unsigned short*)(ws + 134217728);          //  ~38 MiB packed weights
    unsigned short* fcpk  = (unsigned short*)(ws + 174252032);
    float2*         stats = (float2*)(ws + 174268416);
    float*          fcout = (float*)(ws + 174272512);
    float2*         part  = (float2*)(ws + 174282752);                  //  S*C partials

    static const int CH[16] = {64,64,128,128,256,256,256,256,512,512,512,512,512,512,512,512};
    static const int SP[16] = {32,32,16,16,8,8,8,8,4,4,4,4,2,2,2,2};
    static const int PL[16] = {0,1,0,1,0,0,0,1,0,0,0,1,0,0,0,1};
    const int S = 8;   // BN reduction slices

    // repack weights (layers 1..15) and fc
    size_t woff[16]; size_t acc = 0;
    for (int i = 1; i < 16; ++i) {
        woff[i] = acc;
        acc += (size_t)CH[i] * CH[i - 1] * 9;
    }
    for (int i = 1; i < 16; ++i) {
        int Cin = CH[i - 1], Cout = CH[i];
        dim3 grid((Cin * 9 + 255) / 256, Cout);
        repack_w<<<grid, 256, 0, stream>>>(convw[i], wpk + woff[i], Cin);
    }
    repack_fc<<<(16 * 512 + 255) / 256, 256, 0, stream>>>(fcw, fcpk);

    // layer 0: direct conv + BN + ReLU
    conv_first<<<(256 * 1024 * 64) / 256, 256, 0, stream>>>(x, convw[0], convb[0], convout);
    unsigned short* cur = actA;
    unsigned short* nxt = actB;
    {
        int C = 64, Nelem = 256 * 32 * 32;
        bn_partial<<<dim3(C, S), 256, 0, stream>>>(convout, Nelem, C, part);
        bn_finalize<<<(C + 255) / 256, 256, 0, stream>>>(part, S, C, Nelem,
                                                         bng[0], bnb[0], stats);
        size_t tot = (size_t)256 * 32 * 32 * C;
        bn_relu_pool<<<(unsigned)((tot + 255) / 256), 256, 0, stream>>>(
            convout, stats, cur, ilog2(C), ilog2(32), 0);
    }

    // layers 1..15: WMMA conv + BN + ReLU (+pool)
    for (int i = 1; i < 16; ++i) {
        int H = SP[i], C = CH[i], Cin = CH[i - 1];
        launch_conv(H, Cin, C, cur, wpk + woff[i], convb[i], convout, stream);
        int Nelem = 256 * H * H;
        bn_partial<<<dim3(C, S), 256, 0, stream>>>(convout, Nelem, C, part);
        bn_finalize<<<(C + 255) / 256, 256, 0, stream>>>(part, S, C, Nelem,
                                                         bng[i], bnb[i], stats);
        int Ho = PL[i] ? H / 2 : H;
        size_t tot = (size_t)256 * Ho * Ho * C;
        bn_relu_pool<<<(unsigned)((tot + 255) / 256), 256, 0, stream>>>(
            convout, stats, nxt, ilog2(C), ilog2(Ho), PL[i]);
        unsigned short* tmp = cur; cur = nxt; nxt = tmp;
    }

    // FC: cur holds (256,1,1,512) bf16 == flat (256,512)
    fc_wmma<<<2, 256, 0, stream>>>(cur, fcpk, fcb, fcout);

    // bandpass on original input with per-sample weights
    bandpass<<<256 * 3, 1024, 0, stream>>>(x, fcout, (float*)d_out);
}
